// CrossAttentionFusion1_42863773614344
// MI455X (gfx1250) — compile-verified, matrix operand order checked by reading, not run on hardware
//
#include <hip/hip_runtime.h>

// ---------------------------------------------------------------------------
// CrossAttentionFusion for MI455X (gfx1250, wave32, WMMA bf16)
// B=8, N=64, T=128, E=128, H=8, R=4, DH=16  -> 65536 positions
// ---------------------------------------------------------------------------

typedef __attribute__((ext_vector_type(16))) __bf16 v16bf;
typedef __attribute__((ext_vector_type(2)))  __bf16 v2bf;
typedef __attribute__((ext_vector_type(8)))  float  v8f;

#define E_DIM   128
#define R_DIM   4
#define H_DIM   8
#define DH_DIM  16
#define NPOS    (8 * 64 * 128)
#define POS_PER_WG 32
#define NBLOCKS (NPOS / POS_PER_WG)   // 2048

// ---- workspace layout (bytes) ----
#define WQKV_OFF   0                        // bf16 [3][128][128]
#define WQKV_BYTES (3 * 128 * 128 * 2)      // 98304
#define WOB_OFF    (WQKV_OFF + WQKV_BYTES)  // bf16 [128][512]
#define WOB_BYTES  (128 * 512 * 2)          // 131072
#define BQKV_OFF   (WOB_OFF + WOB_BYTES)    // f32 [3][128]
#define BOUT_OFF   (BQKV_OFF + 3 * 128 * 4) // f32 [128]

// ---- LDS layout (bytes) ----
#define PITCHX_B 272                         // 136 bf16, 16B-aligned, conflict-free
#define X_OFF    0
#define X_BYTES  (128 * PITCHX_B)            // 34816
#define QKV_OFF  X_BYTES                     // 3 sections of 34816 (q,k,v)
#define QKV_SECT X_BYTES
#define O_OFF    0                           // aliases X (dead after phase 1)
#define PITCHO_B 1040                        // 520 bf16, 16B-aligned, conflict-free
#define SMEM_BYTES (X_BYTES + 3 * QKV_SECT)  // 139264

// Native bf16 converts (RTNE) -> lets backend use v_cvt_*bf16_f32
__device__ __forceinline__ unsigned short f2bf(float f) {
    return __builtin_bit_cast(unsigned short, (__bf16)f);
}
__device__ __forceinline__ unsigned pack2bf(float lo, float hi) {
    v2bf t;
    t[0] = (__bf16)lo;
    t[1] = (__bf16)hi;
    return __builtin_bit_cast(unsigned, t);
}
__device__ __forceinline__ float bf2f_lo(unsigned u) {
    return __builtin_bit_cast(float, u << 16);
}
__device__ __forceinline__ float bf2f_hi(unsigned u) {
    return __builtin_bit_cast(float, u & 0xFFFF0000u);
}

union FragBF {
    uint4 u[2];
    v16bf v;
};

// A/B fragment per ISA layout: lane L holds row (L%16); elems 0..7 = K[col0..col0+7],
// elems 8..15 = K[col0+16..col0+23], col0 = kt*32 + (L/16)*8.
__device__ __forceinline__ v16bf frag_from(const char* p) {
    FragBF f;
    f.u[0] = *(const uint4*)p;
    f.u[1] = *(const uint4*)(p + 32);
    return f.v;
}

__device__ __forceinline__ void lds_row16(const char* p, float* o) {
    const uint4* q = (const uint4*)p;
    uint4 a = q[0], b = q[1];
    unsigned u[8] = {a.x, a.y, a.z, a.w, b.x, b.y, b.z, b.w};
#pragma unroll
    for (int i = 0; i < 8; ++i) {
        o[2 * i]     = bf2f_lo(u[i]);
        o[2 * i + 1] = bf2f_hi(u[i]);
    }
}

// ---------------------------------------------------------------------------
// Prep: fold in_w into Wq/Wk/Wv (with 1/sqrt(DH) on q), fold out_w into fc_w.
// Outputs bf16 fused weights + f32 fused biases into the workspace.
// ---------------------------------------------------------------------------
__global__ void caf_prep_kernel(const float* __restrict__ Wq, const float* __restrict__ bq,
                                const float* __restrict__ Wk, const float* __restrict__ bk,
                                const float* __restrict__ Wv, const float* __restrict__ bv,
                                const float* __restrict__ in_w, const float* __restrict__ in_b,
                                const float* __restrict__ out_w, const float* __restrict__ out_b,
                                const float* __restrict__ fc_w, const float* __restrict__ fc_b,
                                char* __restrict__ ws) {
    const int T1 = 3 * 128 * 128;   // fused QKV weights
    const int T2 = 128 * 512;       // fused out/fc weight
    const int T3 = 3 * 128;         // fused QKV biases
    const int T4 = 128;             // fused out bias
    unsigned short* wqkv = (unsigned short*)(ws + WQKV_OFF);
    unsigned short* wob  = (unsigned short*)(ws + WOB_OFF);
    float* bqkv = (float*)(ws + BQKV_OFF);
    float* bout = (float*)(ws + BOUT_OFF);

    for (int idx = blockIdx.x * blockDim.x + threadIdx.x; idx < T1 + T2 + T3 + T4;
         idx += gridDim.x * blockDim.x) {
        if (idx < T1) {
            int m = idx / 16384, rem = idx & 16383;
            int o = rem >> 7, f = rem & 127;
            const float* W = (m == 0) ? Wq : (m == 1) ? Wk : Wv;
            const float* irow = in_w + (m * 128 + o) * 128;
            float s = 0.f;
            for (int e = 0; e < 128; ++e) s += irow[e] * W[e * 128 + f];
            if (m == 0) s *= 0.25f;   // 1/sqrt(DH)
            wqkv[idx] = f2bf(s);
        } else if (idx < T1 + T2) {
            int j = idx - T1;
            int e = j >> 9, c = j & 511;
            int r = c >> 7, g = c & 127;
            const float* frow = fc_w + e * 512 + r * 128;
            float s = 0.f;
            for (int f = 0; f < 128; ++f) s += frow[f] * out_w[f * 128 + g];
            wob[j] = f2bf(s);
        } else if (idx < T1 + T2 + T3) {
            int j = idx - T1 - T2;
            int m = j >> 7, o = j & 127;
            const float* bvec = (m == 0) ? bq : (m == 1) ? bk : bv;
            const float* irow = in_w + (m * 128 + o) * 128;
            float s = in_b[m * 128 + o];
            for (int e = 0; e < 128; ++e) s += irow[e] * bvec[e];
            if (m == 0) s *= 0.25f;
            bqkv[j] = s;
        } else {
            int e = idx - T1 - T2 - T3;
            float s = fc_b[e];
            for (int c = 0; c < 512; ++c) s += fc_w[e * 512 + c] * out_b[c & 127];
            bout[e] = s;
        }
    }
}

// ---------------------------------------------------------------------------
// Main fused kernel: 256 threads (8 waves) per WG, 32 positions per WG.
// ---------------------------------------------------------------------------
__global__ void __launch_bounds__(256)
caf_main_kernel(const float* __restrict__ node, const float* __restrict__ subg,
                const char* __restrict__ ws, float* __restrict__ out) {
    extern __shared__ char sm[];

    const int pos0 = blockIdx.x * POS_PER_WG;
    const int tid  = threadIdx.x;
    const int wv   = tid >> 5;          // wave id 0..7
    const int lane = tid & 31;
    const int lrow = lane & 15;         // fragment row / output column within tile
    const int khalf = lane >> 4;        // which K-half this lane holds

    // ---------- Phase 0: stage combined X [128 rows x 128] as bf16 ----------
#pragma unroll 4
    for (int i = tid; i < 128 * 128; i += 256) {
        int row = i >> 7, e = i & 127;
        int lp = row >> 2, r = row & 3;
        int gpos = pos0 + lp;
        float v = (r == 0) ? node[gpos * 128 + e]
                           : subg[(size_t)gpos * 384 + e * 3 + (r - 1)];
        *(unsigned short*)(sm + X_OFF + row * PITCHX_B + e * 2) = f2bf(v);
    }
    __syncthreads();

    // ---------- Phase 1: fused q/k/v projections (WMMA bf16) ----------
    const float* bqkv = (const float*)(ws + BQKV_OFF);
    {
        const char* Xb = sm + X_OFF + (wv * 16 + lrow) * PITCHX_B;
        for (int m = 0; m < 3; ++m) {
            const char* Wm = ws + WQKV_OFF + m * (128 * 128 * 2);
            char* dst = sm + QKV_OFF + m * QKV_SECT;
            for (int ct = 0; ct < 8; ++ct) {
                float b = bqkv[m * 128 + ct * 16 + lrow];
                v8f acc;
#pragma unroll
                for (int g = 0; g < 8; ++g) acc[g] = b;
                const char* Wrow = Wm + (ct * 16 + lrow) * 256;
#pragma unroll
                for (int kt = 0; kt < 4; ++kt) {
                    int col0 = kt * 32 + khalf * 8;
                    v16bf a  = frag_from(Xb + col0 * 2);
                    v16bf bb = frag_from(Wrow + col0 * 2);
                    acc = __builtin_amdgcn_wmma_f32_16x16x32_bf16(
                        false, a, false, bb, (short)0, acc, false, false);
                }
#pragma unroll
                for (int g = 0; g < 8; ++g) {
                    int row = wv * 16 + g + khalf * 8;
                    int col = ct * 16 + lrow;
                    *(unsigned short*)(dst + row * PITCHX_B + col * 2) = f2bf(acc[g]);
                }
            }
        }
    }
    __syncthreads();

    // ---------- Phase 2: 4x4 per-head attention (VALU, tiny FLOP share) ----
    {
        const int lp = tid >> 3;            // local position 0..31
        const int h  = tid & 7;             // head 0..7
        const int rb = lp * 4;              // q/k/v row base
        const char* qb = sm + QKV_OFF;
        const char* kb = sm + QKV_OFF + QKV_SECT;
        const char* vb = sm + QKV_OFF + 2 * QKV_SECT;

        float kk[4][16];
#pragma unroll
        for (int s = 0; s < 4; ++s)
            lds_row16(kb + (rb + s) * PITCHX_B + h * 32, kk[s]);

#pragma unroll
        for (int r = 0; r < 4; ++r) {
            float q[16];
            lds_row16(qb + (rb + r) * PITCHX_B + h * 32, q);
            float sc[4];
#pragma unroll
            for (int s = 0; s < 4; ++s) {
                float d = 0.f;
#pragma unroll
                for (int j = 0; j < 16; ++j) d += q[j] * kk[s][j];
                sc[s] = d;                   // q already scaled by 1/sqrt(DH)
            }
            float mx = fmaxf(fmaxf(sc[0], sc[1]), fmaxf(sc[2], sc[3]));
            float a[4], den = 0.f;
#pragma unroll
            for (int s = 0; s < 4; ++s) { a[s] = __expf(sc[s] - mx); den += a[s]; }
            float inv = 1.f / den;
#pragma unroll
            for (int s = 0; s < 4; ++s) a[s] *= inv;

            float o[16];
#pragma unroll
            for (int j = 0; j < 16; ++j) o[j] = 0.f;
#pragma unroll
            for (int s = 0; s < 4; ++s) {
                float vr[16];
                lds_row16(vb + (rb + s) * PITCHX_B + h * 32, vr);
#pragma unroll
                for (int j = 0; j < 16; ++j) o[j] += a[s] * vr[j];
            }
            unsigned ow[8];
#pragma unroll
            for (int i = 0; i < 8; ++i)
                ow[i] = pack2bf(o[2 * i], o[2 * i + 1]);
            char* od = sm + O_OFF + lp * PITCHO_B + r * 256 + h * 32;
            *(uint4*)od        = make_uint4(ow[0], ow[1], ow[2], ow[3]);
            *(uint4*)(od + 16) = make_uint4(ow[4], ow[5], ow[6], ow[7]);
        }
    }
    __syncthreads();

    // ---------- Phase 3: fused (out_w @ fc_w) GEMM  [32x512] x [512x128] ----
    {
        const float* bout = (const float*)(ws + BOUT_OFF);
#pragma unroll
        for (int i = 0; i < 2; ++i) {
            int tile = wv * 2 + i;
            int mt = tile >> 3, ct = tile & 7;
            float b = bout[ct * 16 + lrow];
            v8f acc;
#pragma unroll
            for (int g = 0; g < 8; ++g) acc[g] = b;
            const char* Ob   = sm + O_OFF + (mt * 16 + lrow) * PITCHO_B;
            const char* Wrow = ws + WOB_OFF + (ct * 16 + lrow) * 1024;
            for (int kt = 0; kt < 16; ++kt) {
                int col0 = kt * 32 + khalf * 8;
                v16bf a  = frag_from(Ob + col0 * 2);
                v16bf bb = frag_from(Wrow + col0 * 2);
                acc = __builtin_amdgcn_wmma_f32_16x16x32_bf16(
                    false, a, false, bb, (short)0, acc, false, false);
            }
#pragma unroll
            for (int g = 0; g < 8; ++g) {
                int gpos = pos0 + mt * 16 + g + 8 * khalf;
                int e    = ct * 16 + lrow;
                out[gpos * 128 + e] = acc[g];
            }
        }
    }
}

// ---------------------------------------------------------------------------
extern "C" void kernel_launch(void* const* d_in, const int* in_sizes, int n_in,
                              void* d_out, int out_size, void* d_ws, size_t ws_size,
                              hipStream_t stream) {
    const float* node  = (const float*)d_in[0];
    const float* subg  = (const float*)d_in[1];
    const float* Wq    = (const float*)d_in[2];
    const float* bq    = (const float*)d_in[3];
    const float* Wk    = (const float*)d_in[4];
    const float* bk    = (const float*)d_in[5];
    const float* Wv    = (const float*)d_in[6];
    const float* bv    = (const float*)d_in[7];
    const float* in_w  = (const float*)d_in[8];
    const float* in_b  = (const float*)d_in[9];
    const float* out_w = (const float*)d_in[10];
    const float* out_b = (const float*)d_in[11];
    const float* fc_w  = (const float*)d_in[12];
    const float* fc_b  = (const float*)d_in[13];
    char* ws   = (char*)d_ws;
    float* out = (float*)d_out;

    caf_prep_kernel<<<dim3(64), dim3(256), 0, stream>>>(
        Wq, bq, Wk, bk, Wv, bv, in_w, in_b, out_w, out_b, fc_w, fc_b, ws);

    caf_main_kernel<<<dim3(NBLOCKS), dim3(256), SMEM_BYTES, stream>>>(
        node, subg, ws, out);
}
